// NonLocalAttention_81484119540135
// MI455X (gfx1250) — matrix operand order repaired and checked by reading, not compiled, not run on hardware
//
#include <hip/hip_runtime.h>
#include <hip/hip_bf16.h>
#include <stdint.h>

// Problem dims (fixed by reference)
#define BB 8
#define CC 256
#define CI 128
#define NN 4096   // H*W = 64*64

typedef __attribute__((ext_vector_type(16))) __bf16 bf16x16;
typedef __attribute__((ext_vector_type(8)))  float  f32x8;

union Frag {
    bf16x16        v;
    unsigned short s[16];
    uint4          q[2];
};

__device__ __forceinline__ unsigned short f2bf(float f) {
    union { float f; unsigned int u; } a; a.f = f;
    unsigned int u = a.u;
    // round-to-nearest-even fp32 -> bf16
    return (unsigned short)((u + 0x7FFFu + ((u >> 16) & 1u)) >> 16);
}

// ---------------------------------------------------------------------------
// Kernel 0a: fp32 -> bf16 weight conversion
// ---------------------------------------------------------------------------
__global__ void cvt_f32_bf16(const float* __restrict__ src,
                             unsigned short* __restrict__ dst, int n) {
    int i = blockIdx.x * 256 + threadIdx.x;
    if (i < n) dst[i] = f2bf(src[i]);
}

// ---------------------------------------------------------------------------
// Kernel 0b: x [B][C][N] fp32 -> Xt [B][N][C] bf16 (LDS tiled transpose)
// ---------------------------------------------------------------------------
__global__ void __launch_bounds__(256) transpose_x(const float* __restrict__ x,
                                                   unsigned short* __restrict__ xt) {
    __shared__ float tile[32][33];
    const int b  = blockIdx.z;
    const int n0 = blockIdx.x * 32, c0 = blockIdx.y * 32;
    const int tx = threadIdx.x, ty = threadIdx.y;   // (32, 8)
    #pragma unroll
    for (int r = 0; r < 32; r += 8)
        tile[ty + r][tx] = x[((size_t)b * CC + (c0 + ty + r)) * NN + n0 + tx];
    __syncthreads();
    #pragma unroll
    for (int r = 0; r < 32; r += 8)
        xt[((size_t)b * NN + (n0 + ty + r)) * CC + c0 + tx] = f2bf(tile[tx][ty + r]);
}

// ---------------------------------------------------------------------------
// Kernel 1: QKV projection.
//   Qt,Kt [B][N][CI] bf16 (row = n)    Vm [B][CI][N] bf16 (row = ci)
// One wave per (16-n x 16-ci) tile; 8 waves/block cover all CI for one n-tile.
// ---------------------------------------------------------------------------
__global__ void __launch_bounds__(256) qkv_proj(
    const unsigned short* __restrict__ xt,
    const unsigned short* __restrict__ wqb, const float* __restrict__ bq,
    const unsigned short* __restrict__ wkb, const float* __restrict__ bk,
    const unsigned short* __restrict__ wvb, const float* __restrict__ bv,
    unsigned short* __restrict__ qt, unsigned short* __restrict__ kt,
    unsigned short* __restrict__ vm)
{
    const int lane = threadIdx.x & 31, w = threadIdx.x >> 5;
    const int g = lane >> 4, l = lane & 15;
    const int b = blockIdx.y, n0 = blockIdx.x * 16, ci0 = w * 16;

    const unsigned short* xrow  = xt  + ((size_t)b * NN + n0 + l) * CC; // A rows (n)
    const unsigned short* wvrow = wvb + (size_t)(ci0 + l) * CC;         // A rows (ci)
    const unsigned short* wqrow = wqb + (size_t)(ci0 + l) * CC;         // B cols (ci)
    const unsigned short* wkrow = wkb + (size_t)(ci0 + l) * CC;

    f32x8 dq = {}, dk = {}, dv = {};
    #pragma unroll
    for (int k = 0; k < 8; ++k) {
        Frag ax, av, bwq, bwk, bx;
        ax.q[0]  = *(const uint4*)(xrow  + k * 32 + g * 16);
        ax.q[1]  = *(const uint4*)(xrow  + k * 32 + g * 16 + 8);
        av.q[0]  = *(const uint4*)(wvrow + k * 32 + g * 16);
        av.q[1]  = *(const uint4*)(wvrow + k * 32 + g * 16 + 8);
        bwq.q[0] = *(const uint4*)(wqrow + k * 32 + g * 16);
        bwq.q[1] = *(const uint4*)(wqrow + k * 32 + g * 16 + 8);
        bwk.q[0] = *(const uint4*)(wkrow + k * 32 + g * 16);
        bwk.q[1] = *(const uint4*)(wkrow + k * 32 + g * 16 + 8);
        bx.q[0]  = *(const uint4*)(xrow  + k * 32 + g * 16);
        bx.q[1]  = *(const uint4*)(xrow  + k * 32 + g * 16 + 8);
        dq = __builtin_amdgcn_wmma_f32_16x16x32_bf16(false, ax.v, false, bwq.v, (short)0, dq, false, false);
        dk = __builtin_amdgcn_wmma_f32_16x16x32_bf16(false, ax.v, false, bwk.v, (short)0, dk, false, false);
        dv = __builtin_amdgcn_wmma_f32_16x16x32_bf16(false, av.v, false, bx.v,  (short)0, dv, false, false);
    }
    const float bqv = bq[ci0 + l];
    const float bkv = bk[ci0 + l];
    #pragma unroll
    for (int v = 0; v < 8; ++v) {
        const int n  = n0 + v + 8 * g;      // Q/K: rows = n, cols = ci
        const int cv = ci0 + v + 8 * g;     // V : rows = ci, cols = n
        qt[((size_t)b * NN + n) * CI + ci0 + l] = f2bf(dq[v] + bqv);
        kt[((size_t)b * NN + n) * CI + ci0 + l] = f2bf(dk[v] + bkv);
        vm[((size_t)b * CI + cv) * NN + n0 + l] = f2bf(dv[v] + bv[cv]);
    }
}

// ---------------------------------------------------------------------------
// Kernel 2: fused flash attention. One wave handles 16 query rows, streams all
// of K/V for its batch in j-tiles of 32 with online softmax.
// o2 [B][N][CI] bf16.
// ---------------------------------------------------------------------------
__global__ void __launch_bounds__(128) attn_fused(
    const unsigned short* __restrict__ qt,
    const unsigned short* __restrict__ kt,
    const unsigned short* __restrict__ vm,
    unsigned short* __restrict__ o2)
{
    __shared__ __align__(16) unsigned short plds[4][16 * 32];   // per-wave P tile
    const int lane = threadIdx.x & 31, w = threadIdx.x >> 5;
    const int g = lane >> 4, l = lane & 15;
    const int b  = blockIdx.y;
    const int i0 = blockIdx.x * 64 + w * 16;

    // Q fragments (A layout, rows = i) resident across the whole j loop
    Frag aq[4];
    const unsigned short* qrow = qt + ((size_t)b * NN + i0 + l) * CI;
    #pragma unroll
    for (int k = 0; k < 4; ++k) {
        aq[k].q[0] = *(const uint4*)(qrow + k * 32 + g * 16);
        aq[k].q[1] = *(const uint4*)(qrow + k * 32 + g * 16 + 8);
    }

    f32x8 acc[8];
    #pragma unroll
    for (int t = 0; t < 8; ++t) acc[t] = (f32x8){};
    float m[8], lsum[8];
    #pragma unroll
    for (int v = 0; v < 8; ++v) { m[v] = -3.0e38f; lsum[v] = 0.0f; }

    const unsigned short* ktb = kt + (size_t)b * NN * CI;
    const unsigned short* vmb = vm + (size_t)b * CI * NN;
    unsigned short* myp = plds[w];

    for (int j0 = 0; j0 < NN; j0 += 32) {
        if (j0 + 32 < NN)   // stream next K tile toward the caches
            __builtin_prefetch((const void*)(ktb + (size_t)(j0 + 32 + l) * CI), 0, 1);

        // S = Q^T K for two 16-wide j sub-tiles
        f32x8 s0 = {}, s1 = {};
        const unsigned short* k0p = ktb + (size_t)(j0 + l) * CI + g * 16;
        const unsigned short* k1p = ktb + (size_t)(j0 + 16 + l) * CI + g * 16;
        #pragma unroll
        for (int k = 0; k < 4; ++k) {
            Frag bk0, bk1;
            bk0.q[0] = *(const uint4*)(k0p + k * 32);
            bk0.q[1] = *(const uint4*)(k0p + k * 32 + 8);
            bk1.q[0] = *(const uint4*)(k1p + k * 32);
            bk1.q[1] = *(const uint4*)(k1p + k * 32 + 8);
            s0 = __builtin_amdgcn_wmma_f32_16x16x32_bf16(false, aq[k].v, false, bk0.v, (short)0, s0, false, false);
            s1 = __builtin_amdgcn_wmma_f32_16x16x32_bf16(false, aq[k].v, false, bk1.v, (short)0, s1, false, false);
        }

        // Online softmax. D layout: lane holds col j=l (s0) / 16+l (s1),
        // register v holds row r = v + 8*g. Row reductions across 16 lanes.
        #pragma unroll
        for (int v = 0; v < 8; ++v) {
            float tmax = fmaxf(s0[v], s1[v]);
            #pragma unroll
            for (int msk = 1; msk < 16; msk <<= 1)
                tmax = fmaxf(tmax, __shfl_xor(tmax, msk, 32));
            const float mn = fmaxf(m[v], tmax);
            const float sc = __expf(m[v] - mn);
            const float p0 = __expf(s0[v] - mn);
            const float p1 = __expf(s1[v] - mn);
            float rs = p0 + p1;
            #pragma unroll
            for (int msk = 1; msk < 16; msk <<= 1)
                rs += __shfl_xor(rs, msk, 32);
            lsum[v] = lsum[v] * sc + rs;
            m[v] = mn;
            #pragma unroll
            for (int t = 0; t < 8; ++t) acc[t][v] *= sc;   // rescale out accum
            const int r = v + 8 * g;
            myp[r * 32 + l]      = f2bf(p0);               // D -> LDS (row-major 16x32)
            myp[r * 32 + 16 + l] = f2bf(p1);
        }
        // Cross-lane LDS dependency: same-wave DS ops are in order, but stop
        // the compiler reordering and drain DScnt before the b128 reads.
        __asm__ volatile("s_wait_dscnt 0" ::: "memory");

        // Re-read P in A layout (rows = i, K = 32 j values)
        Frag ap;
        ap.q[0] = *(const uint4*)(myp + l * 32 + g * 16);
        ap.q[1] = *(const uint4*)(myp + l * 32 + g * 16 + 8);

        // acc += P * V^T across the 8 c-tiles (B layout: lane=col c, K=j)
        #pragma unroll
        for (int t = 0; t < 8; ++t) {
            Frag bv_;
            const unsigned short* p = vmb + (size_t)(t * 16 + l) * NN + j0 + g * 16;
            bv_.q[0] = *(const uint4*)(p);
            bv_.q[1] = *(const uint4*)(p + 8);
            acc[t] = __builtin_amdgcn_wmma_f32_16x16x32_bf16(false, ap.v, false, bv_.v, (short)0, acc[t], false, false);
        }
        __asm__ volatile("" ::: "memory");   // keep LDS reads before next overwrite
    }

    // Normalize rows by softmax denominator and store bf16 [N][CI]
    #pragma unroll
    for (int v = 0; v < 8; ++v) {
        const float inv = 1.0f / lsum[v];
        const size_t row = ((size_t)b * NN + i0 + v + 8 * g) * CI;
        #pragma unroll
        for (int t = 0; t < 8; ++t)
            o2[row + t * 16 + l] = f2bf(acc[t][v] * inv);
    }
}

// ---------------------------------------------------------------------------
// Kernel 3: output projection + gamma residual, fp32 out [B][C][N]
// ---------------------------------------------------------------------------
__global__ void __launch_bounds__(128) out_proj(
    const unsigned short* __restrict__ o2,
    const unsigned short* __restrict__ wob,
    const float* __restrict__ bo,
    const float* __restrict__ gamma,
    const float* __restrict__ x,
    float* __restrict__ out)
{
    const int lane = threadIdx.x & 31, w = threadIdx.x >> 5;
    const int g = lane >> 4, l = lane & 15;
    const int b   = blockIdx.z;
    const int n0  = blockIdx.x * 16;
    const int co0 = blockIdx.y * 64 + w * 16;

    const unsigned short* orow = o2  + ((size_t)b * NN + n0 + l) * CI;
    const unsigned short* wrow = wob + (size_t)(co0 + l) * CI;
    f32x8 d = {};
    #pragma unroll
    for (int k = 0; k < 4; ++k) {
        Frag a, bf;
        a.q[0]  = *(const uint4*)(orow + k * 32 + g * 16);
        a.q[1]  = *(const uint4*)(orow + k * 32 + g * 16 + 8);
        bf.q[0] = *(const uint4*)(wrow + k * 32 + g * 16);
        bf.q[1] = *(const uint4*)(wrow + k * 32 + g * 16 + 8);
        d = __builtin_amdgcn_wmma_f32_16x16x32_bf16(false, a.v, false, bf.v, (short)0, d, false, false);
    }
    const float bias = bo[co0 + l];
    const float gm   = gamma[0];
    const size_t base = ((size_t)b * CC + co0 + l) * NN;
    #pragma unroll
    for (int v = 0; v < 8; ++v) {
        const size_t idx = base + n0 + v + 8 * g;
        out[idx] = gm * (d[v] + bias) + x[idx];
    }
}

// ---------------------------------------------------------------------------
extern "C" void kernel_launch(void* const* d_in, const int* in_sizes, int n_in,
                              void* d_out, int out_size, void* d_ws, size_t ws_size,
                              hipStream_t stream)
{
    (void)in_sizes; (void)n_in; (void)out_size; (void)ws_size;
    const float* x     = (const float*)d_in[0];
    const float* wq    = (const float*)d_in[1];
    const float* bq    = (const float*)d_in[2];
    const float* wk    = (const float*)d_in[3];
    const float* bk    = (const float*)d_in[4];
    const float* wv    = (const float*)d_in[5];
    const float* bv    = (const float*)d_in[6];
    const float* wo    = (const float*)d_in[7];
    const float* bo    = (const float*)d_in[8];
    const float* gamma = (const float*)d_in[9];
    float* out = (float*)d_out;

    char* ws = (char*)d_ws;
    size_t off = 0;
    auto carve = [&](size_t bytes) {
        char* p = ws + off;
        off += (bytes + 255) & ~(size_t)255;
        return p;
    };
    unsigned short* xt  = (unsigned short*)carve((size_t)BB * NN * CC * 2); // 16 MB
    unsigned short* qt  = (unsigned short*)carve((size_t)BB * NN * CI * 2); //  8 MB
    unsigned short* kt  = (unsigned short*)carve((size_t)BB * NN * CI * 2);
    unsigned short* vm  = (unsigned short*)carve((size_t)BB * CI * NN * 2);
    unsigned short* o2  = (unsigned short*)carve((size_t)BB * NN * CI * 2);
    unsigned short* wqb = (unsigned short*)carve((size_t)CI * CC * 2);
    unsigned short* wkb = (unsigned short*)carve((size_t)CI * CC * 2);
    unsigned short* wvb = (unsigned short*)carve((size_t)CI * CC * 2);
    unsigned short* wob = (unsigned short*)carve((size_t)CC * CI * 2);

    const int nw = CI * CC;
    cvt_f32_bf16<<<(nw + 255) / 256, 256, 0, stream>>>(wq, wqb, nw);
    cvt_f32_bf16<<<(nw + 255) / 256, 256, 0, stream>>>(wk, wkb, nw);
    cvt_f32_bf16<<<(nw + 255) / 256, 256, 0, stream>>>(wv, wvb, nw);
    cvt_f32_bf16<<<(nw + 255) / 256, 256, 0, stream>>>(wo, wob, nw);

    transpose_x<<<dim3(NN / 32, CC / 32, BB), dim3(32, 8), 0, stream>>>(x, xt);

    qkv_proj<<<dim3(NN / 16, BB), 256, 0, stream>>>(xt, wqb, bq, wkb, bk, wvb, bv,
                                                    qt, kt, vm);

    attn_fused<<<dim3(NN / 64, BB), 128, 0, stream>>>(qt, kt, vm, o2);

    out_proj<<<dim3(NN / 16, CC / 64, BB), 128, 0, stream>>>(o2, wob, bo, gamma, x, out);
}
